// RQVae_3599182594532
// MI455X (gfx1250) — compile-verified, matrix operand order checked by reading, not run on hardware
//
#include <hip/hip_runtime.h>
#include <hip/hip_bf16.h>
#include <cfloat>

typedef __bf16 bf16_t;
typedef __attribute__((ext_vector_type(16))) __bf16 v16bf;
typedef __attribute__((ext_vector_type(8)))  __bf16 v8bf;
typedef __attribute__((ext_vector_type(8)))  float  v8f;
typedef __attribute__((ext_vector_type(4)))  int    v4i;

#define N_ROWS   65536
#define D_IN     1024
#define D_H      512
#define D_Z      256

#define KBLK     64            // k-chunk staged in LDS per iteration
#define KS       72            // padded LDS row stride (elements) to spread banks
#define LDSB_ELEMS (2 * 64 * KS)

#ifndef __has_builtin
#define __has_builtin(x) 0
#endif
#if __has_builtin(__builtin_amdgcn_global_load_async_to_lds_b128)
#define HAVE_ASYNC_LDS 1
#else
#define HAVE_ASYNC_LDS 0
#endif

// ---------------- helpers ----------------

__device__ inline float gelu_exact(float v) {
    return 0.5f * v * (1.0f + erff(v * 0.70710678118654752f));
}

// 16x32 bf16 fragment (MxK), row-major source, leading dim ld (elements).
// lane L: m = L&15, hi = L>>4; halves 0..7 at k = hi*8+j, 8..15 at k = 16+hi*8+j
// (CDNA5 ISA 7.12.2 16-bit A/B layout; B uses pre-transposed [n][k] storage).
__device__ inline v16bf frag_bf16(const bf16_t* base, int ld) {
    int lane = threadIdx.x & 31;
    const bf16_t* p = base + (size_t)(lane & 15) * ld + ((lane >> 4) << 3);
    v8bf lo = *reinterpret_cast<const v8bf*>(p);
    v8bf hi = *reinterpret_cast<const v8bf*>(p + 16);
    v16bf r;
#pragma unroll
    for (int i = 0; i < 8; i++) { r[i] = lo[i]; r[i + 8] = hi[i]; }
    return r;
}

__device__ inline v16bf frag_f32(const float* base, int ld) {
    int lane = threadIdx.x & 31;
    const float* p = base + (size_t)(lane & 15) * ld + ((lane >> 4) << 3);
    v16bf r;
#pragma unroll
    for (int i = 0; i < 8; i++) { r[i] = (bf16_t)p[i]; r[i + 8] = (bf16_t)p[16 + i]; }
    return r;
}

__device__ inline unsigned long long shfl_xor_u64(unsigned long long v, int mask) {
    unsigned int lo = __shfl_xor((unsigned int)(v & 0xffffffffull), mask, 32);
    unsigned int hi = __shfl_xor((unsigned int)(v >> 32), mask, 32);
    return ((unsigned long long)hi << 32) | lo;
}

// ---- cooperative B-tile staging: 64 n-rows x 64 k (bf16), 32B per thread ----

#if HAVE_ASYNC_LDS
__device__ inline void async_b128(const bf16_t* g, bf16_t* l) {
    // builtin signature: (v4i AS1* src, v4i AS3* dst, imm offset, imm cpol)
    v4i __attribute__((address_space(1)))* gp =
        (v4i __attribute__((address_space(1)))*)(uintptr_t)g;
    v4i __attribute__((address_space(3)))* lp =
        (v4i __attribute__((address_space(3)))*)(unsigned int)(uintptr_t)l;
    __builtin_amdgcn_global_load_async_to_lds_b128(gp, lp, 0, 0);
}
#endif

__device__ inline void tile_issue(const bf16_t* g, bf16_t* l, uint4& ra, uint4& rb) {
#if HAVE_ASYNC_LDS
    async_b128(g, l);
    async_b128(g + 8, l + 8);
    (void)ra; (void)rb;
#else
    ra = *reinterpret_cast<const uint4*>(g);
    rb = *reinterpret_cast<const uint4*>(g + 8);
    (void)l;
#endif
}

__device__ inline void tile_commit(bf16_t* l, const uint4& ra, const uint4& rb) {
#if !HAVE_ASYNC_LDS
    *reinterpret_cast<uint4*>(l) = ra;
    *reinterpret_cast<uint4*>(l + 8) = rb;
#else
    (void)l; (void)ra; (void)rb;
#endif
}

__device__ inline void tile_wait() {
#if HAVE_ASYNC_LDS
#if __has_builtin(__builtin_amdgcn_s_wait_asynccnt)
    __builtin_amdgcn_s_wait_asynccnt(0);
#else
    asm volatile("s_wait_asynccnt 0x0" ::: "memory");
#endif
#endif
}

// ---- staged WMMA cores: one wave computes a 32x64 strip (acc[2][4]) ----
// All four B fragments + both A fragments are loaded at the top of each
// ks-step so DS/VMEM latency is covered by one clause + partial waits.
// Kernels carry __launch_bounds__(256) so the VGPR budget (~2 waves/SIMD)
// accommodates this working set without scratch spills.

#define MMA_CORE_BODY(FRAG_A)                                                     \
    int tid = threadIdx.x;                                                        \
    int cn = tid >> 2;                                                            \
    int cp = (tid & 3) * 16;                                                      \
    const bf16_t* gB = Bt + (size_t)cn * ldb + cp;                                \
    bf16_t* lB = ldsB + cn * KS + cp;                                             \
    uint4 ra, rb;                                                                 \
    /* preload tile 0 */                                                          \
    tile_issue(gB, lB, ra, rb);                                                   \
    tile_commit(lB, ra, rb);                                                      \
    tile_wait();                                                                  \
    __syncthreads();                                                              \
    int buf = 0;                                                                  \
    for (int kb = 0; kb < K; kb += KBLK) {                                        \
        bool next = (kb + KBLK) < K;                                              \
        if (next)                                                                 \
            tile_issue(gB + kb + KBLK, lB + (buf ^ 1) * (64 * KS), ra, rb);       \
        if (next) __builtin_prefetch(A + kb + 2 * KBLK, 0, 1);                    \
        const bf16_t* LB = ldsB + buf * (64 * KS);                                \
        _Pragma("unroll")                                                         \
        for (int ks = 0; ks < KBLK; ks += 32) {                                   \
            v16bf a0 = FRAG_A(A + kb + ks, lda);                                  \
            v16bf a1 = FRAG_A(A + (size_t)16 * lda + kb + ks, lda);               \
            v16bf b0 = frag_bf16(LB + 0 * 16 * KS + ks, KS);                      \
            v16bf b1 = frag_bf16(LB + 1 * 16 * KS + ks, KS);                      \
            v16bf b2 = frag_bf16(LB + 2 * 16 * KS + ks, KS);                      \
            v16bf b3 = frag_bf16(LB + 3 * 16 * KS + ks, KS);                      \
            acc[0][0] = __builtin_amdgcn_wmma_f32_16x16x32_bf16(                  \
                false, a0, false, b0, (short)0, acc[0][0], false, false);         \
            acc[1][0] = __builtin_amdgcn_wmma_f32_16x16x32_bf16(                  \
                false, a1, false, b0, (short)0, acc[1][0], false, false);         \
            acc[0][1] = __builtin_amdgcn_wmma_f32_16x16x32_bf16(                  \
                false, a0, false, b1, (short)0, acc[0][1], false, false);         \
            acc[1][1] = __builtin_amdgcn_wmma_f32_16x16x32_bf16(                  \
                false, a1, false, b1, (short)0, acc[1][1], false, false);         \
            acc[0][2] = __builtin_amdgcn_wmma_f32_16x16x32_bf16(                  \
                false, a0, false, b2, (short)0, acc[0][2], false, false);         \
            acc[1][2] = __builtin_amdgcn_wmma_f32_16x16x32_bf16(                  \
                false, a1, false, b2, (short)0, acc[1][2], false, false);         \
            acc[0][3] = __builtin_amdgcn_wmma_f32_16x16x32_bf16(                  \
                false, a0, false, b3, (short)0, acc[0][3], false, false);         \
            acc[1][3] = __builtin_amdgcn_wmma_f32_16x16x32_bf16(                  \
                false, a1, false, b3, (short)0, acc[1][3], false, false);         \
        }                                                                         \
        if (next) {                                                               \
            tile_commit(lB + (buf ^ 1) * (64 * KS), ra, rb);                      \
            tile_wait();                                                          \
        }                                                                         \
        __syncthreads();                                                          \
        buf ^= 1;                                                                 \
    }

__device__ inline void mma_core_bf16(const bf16_t* A, int lda,
                                     const bf16_t* Bt, int ldb, int K,
                                     bf16_t* ldsB, v8f acc[2][4]) {
    MMA_CORE_BODY(frag_bf16)
}

__device__ inline void mma_core_f32A(const float* A, int lda,
                                     const bf16_t* Bt, int ldb, int K,
                                     bf16_t* ldsB, v8f acc[2][4]) {
    MMA_CORE_BODY(frag_f32)
}

__device__ inline void acc_zero(v8f acc[2][4]) {
#pragma unroll
    for (int g = 0; g < 2; g++)
#pragma unroll
        for (int t = 0; t < 4; t++)
#pragma unroll
            for (int i = 0; i < 8; i++) acc[g][t][i] = 0.0f;
}

// ---------------- conversion / setup kernels ----------------

__global__ void k_cvt_transpose(const float* __restrict__ W, bf16_t* __restrict__ Wt,
                                int K, int N) {
    long i = (long)blockIdx.x * blockDim.x + threadIdx.x;
    if (i < (long)K * N) {
        int k = (int)(i / N), n = (int)(i % N);
        Wt[(size_t)n * K + k] = (bf16_t)W[i];
    }
}

__global__ void k_cvt_copy(const float* __restrict__ S, bf16_t* __restrict__ D, long n) {
    long i = (long)blockIdx.x * blockDim.x + threadIdx.x;
    if (i < n) D[i] = (bf16_t)S[i];
}

__global__ void k_cnorm(const float* __restrict__ C, float* __restrict__ cn) {
    __shared__ float s[256];
    int d = threadIdx.x;
    float v = C[(size_t)blockIdx.x * 256 + d];
    s[d] = v * v;
    __syncthreads();
    for (int st = 128; st > 0; st >>= 1) {
        if (d < st) s[d] += s[d + st];
        __syncthreads();
    }
    if (d == 0) cn[blockIdx.x] = s[0];
}

__global__ void k_fill_f32(float* p, float v, size_t n) {
    size_t i = (size_t)blockIdx.x * blockDim.x + threadIdx.x;
    if (i < n) p[i] = v;
}

__global__ void k_fill_u64(unsigned long long* p, unsigned long long v, size_t n) {
    size_t i = (size_t)blockIdx.x * blockDim.x + threadIdx.x;
    if (i < n) p[i] = v;
}

// ---------------- GEMM kernels: 8 waves/block, wave = 32x64 strip ----------------
// block covers 256 rows x 64 cols; grid = (M/256, N/64)

__global__ void __launch_bounds__(256)
k_gemm_f32A_bias_gelu(const float* __restrict__ A, int lda,
                      const bf16_t* __restrict__ Bt, int ldb,
                      const float* __restrict__ bias,
                      bf16_t* __restrict__ Out, int ldo, int K) {
    __shared__ bf16_t ldsB[LDSB_ELEMS];
    int wave = threadIdx.x >> 5;
    int row0 = (blockIdx.x * 8 + wave) * 32;
    int n0 = blockIdx.y * 64;
    v8f acc[2][4];
    acc_zero(acc);
    mma_core_f32A(A + (size_t)row0 * lda, lda, Bt + (size_t)n0 * ldb, ldb, K, ldsB, acc);
    int lane = threadIdx.x & 31, li = lane & 15, hi = lane >> 4;
#pragma unroll
    for (int g = 0; g < 2; g++)
#pragma unroll
        for (int t = 0; t < 4; t++) {
            int n = n0 + t * 16 + li;
            float b = bias[n];
#pragma unroll
            for (int r = 0; r < 8; r++) {
                int m = row0 + g * 16 + hi * 8 + r;
                Out[(size_t)m * ldo + n] = (bf16_t)gelu_exact(acc[g][t][r] + b);
            }
        }
}

__global__ void __launch_bounds__(256)
k_gemm_bias_gelu(const bf16_t* __restrict__ A, int lda,
                 const bf16_t* __restrict__ Bt, int ldb,
                 const float* __restrict__ bias,
                 bf16_t* __restrict__ Out, int ldo, int K) {
    __shared__ bf16_t ldsB[LDSB_ELEMS];
    int wave = threadIdx.x >> 5;
    int row0 = (blockIdx.x * 8 + wave) * 32;
    int n0 = blockIdx.y * 64;
    v8f acc[2][4];
    acc_zero(acc);
    mma_core_bf16(A + (size_t)row0 * lda, lda, Bt + (size_t)n0 * ldb, ldb, K, ldsB, acc);
    int lane = threadIdx.x & 31, li = lane & 15, hi = lane >> 4;
#pragma unroll
    for (int g = 0; g < 2; g++)
#pragma unroll
        for (int t = 0; t < 4; t++) {
            int n = n0 + t * 16 + li;
            float b = bias[n];
#pragma unroll
            for (int r = 0; r < 8; r++) {
                int m = row0 + g * 16 + hi * 8 + r;
                Out[(size_t)m * ldo + n] = (bf16_t)gelu_exact(acc[g][t][r] + b);
            }
        }
}

__global__ void __launch_bounds__(256)
k_gemm_bias_dual(const bf16_t* __restrict__ A, int lda,
                 const bf16_t* __restrict__ Bt, int ldb,
                 const float* __restrict__ bias,
                 float* __restrict__ Zf, bf16_t* __restrict__ Zb,
                 int ldo, int K) {
    __shared__ bf16_t ldsB[LDSB_ELEMS];
    int wave = threadIdx.x >> 5;
    int row0 = (blockIdx.x * 8 + wave) * 32;
    int n0 = blockIdx.y * 64;
    v8f acc[2][4];
    acc_zero(acc);
    mma_core_bf16(A + (size_t)row0 * lda, lda, Bt + (size_t)n0 * ldb, ldb, K, ldsB, acc);
    int lane = threadIdx.x & 31, li = lane & 15, hi = lane >> 4;
#pragma unroll
    for (int g = 0; g < 2; g++)
#pragma unroll
        for (int t = 0; t < 4; t++) {
            int n = n0 + t * 16 + li;
            float b = bias[n];
#pragma unroll
            for (int r = 0; r < 8; r++) {
                int m = row0 + g * 16 + hi * 8 + r;
                float v = acc[g][t][r] + b;
                Zf[(size_t)m * ldo + n] = v;
                Zb[(size_t)m * ldo + n] = (bf16_t)v;
            }
        }
}

// Score GEMM + per-row argmin: score = ||C_n||^2 - 2 (r . C_n), min kept as
// atomicMin on u64 = (orderedFloat(score) << 32) | code_index.
__global__ void __launch_bounds__(256)
k_score_argmin(const bf16_t* __restrict__ R, int lda,
               const bf16_t* __restrict__ Cb, int ldb,
               const float* __restrict__ cnorm,
               unsigned long long* __restrict__ minbuf, int K) {
    __shared__ bf16_t ldsB[LDSB_ELEMS];
    int wave = threadIdx.x >> 5;
    int row0 = (blockIdx.x * 8 + wave) * 32;
    int n0 = blockIdx.y * 64;
    v8f acc[2][4];
    acc_zero(acc);
    mma_core_bf16(R + (size_t)row0 * lda, lda, Cb + (size_t)n0 * ldb, ldb, K, ldsB, acc);
    int lane = threadIdx.x & 31, li = lane & 15, hi = lane >> 4;
#pragma unroll
    for (int g = 0; g < 2; g++)
#pragma unroll
        for (int r = 0; r < 8; r++) {
            float best = FLT_MAX;
            int bestn = 0;
#pragma unroll
            for (int t = 0; t < 4; t++) {
                int n = n0 + t * 16 + li;
                float s = cnorm[n] - 2.0f * acc[g][t][r];
                if (s < best) { best = s; bestn = n; }
            }
            unsigned int u = __float_as_uint(best);
            u = (u & 0x80000000u) ? ~u : (u | 0x80000000u);
            unsigned long long pack = ((unsigned long long)u << 32) | (unsigned int)bestn;
#pragma unroll
            for (int msk = 1; msk < 16; msk <<= 1) {
                unsigned long long o = shfl_xor_u64(pack, msk);
                if (o < pack) pack = o;
            }
            if (li == 0) {
                int m = row0 + g * 16 + hi * 8 + r;
                atomicMin(minbuf + m, pack);
            }
        }
}

// Decoder GEMM2 + fused recon loss: sum((A@W + bias - x)^2), nothing stored.
__global__ void __launch_bounds__(256)
k_gemm_recon(const bf16_t* __restrict__ A, int lda,
             const bf16_t* __restrict__ Bt, int ldb,
             const float* __restrict__ bias,
             const float* __restrict__ x, int ldx,
             float* __restrict__ loss_slot, int K) {
    __shared__ bf16_t ldsB[LDSB_ELEMS];
    int wave = threadIdx.x >> 5;
    int row0 = (blockIdx.x * 8 + wave) * 32;
    int n0 = blockIdx.y * 64;
    v8f acc[2][4];
    acc_zero(acc);
    mma_core_bf16(A + (size_t)row0 * lda, lda, Bt + (size_t)n0 * ldb, ldb, K, ldsB, acc);
    int lane = threadIdx.x & 31, li = lane & 15, hi = lane >> 4;
    float lsum = 0.0f;
#pragma unroll
    for (int g = 0; g < 2; g++)
#pragma unroll
        for (int t = 0; t < 4; t++) {
            int n = n0 + t * 16 + li;
            float b = bias[n];
#pragma unroll
            for (int r = 0; r < 8; r++) {
                int m = row0 + g * 16 + hi * 8 + r;
                float d = acc[g][t][r] + b - x[(size_t)m * ldx + n];
                lsum += d * d;
            }
        }
#pragma unroll
    for (int msk = 16; msk > 0; msk >>= 1) lsum += __shfl_xor(lsum, msk, 32);
    if (lane == 0) atomicAdd(loss_slot, lsum);
}

// Quantize stage: q = C[idx]; residual -= q; outq += q; accumulate sum((r-q)^2).
__global__ void __launch_bounds__(256)
k_quant_update(const float* __restrict__ Cf,
               const unsigned long long* __restrict__ minbuf,
               float* __restrict__ residual, bf16_t* __restrict__ rbf,
               float* __restrict__ outq_f, bf16_t* __restrict__ outq_b,
               float* __restrict__ loss_slot) {
    __shared__ float s[256];
    int n = blockIdx.x, d = threadIdx.x;
    unsigned int idx = (unsigned int)(minbuf[n] & 0xffffffffull);
    float q = Cf[(size_t)idx * 256 + d];
    size_t o = (size_t)n * 256 + d;
    float diff = residual[o] - q;
    residual[o] = diff;
    rbf[o] = (bf16_t)diff;
    float oq = outq_f[o] + q;
    outq_f[o] = oq;
    outq_b[o] = (bf16_t)oq;
    s[d] = diff * diff;
    __syncthreads();
    for (int st = 128; st > 0; st >>= 1) {
        if (d < st) s[d] += s[d + st];
        __syncthreads();
    }
    if (d == 0) atomicAdd(loss_slot, s[0]);
}

__global__ void k_finalize(const float* __restrict__ loss, float* __restrict__ out) {
    if (threadIdx.x == 0 && blockIdx.x == 0) {
        out[0] = loss[3] / ((float)N_ROWS * (float)D_IN);
        out[1] = 1.25f * loss[0] / ((float)N_ROWS * (float)D_Z);
        out[2] = 1.25f * loss[1] / ((float)N_ROWS * (float)D_Z);
        out[3] = 1.25f * loss[2] / ((float)N_ROWS * (float)D_Z);
    }
}

// ---------------- launcher ----------------

extern "C" void kernel_launch(void* const* d_in, const int* in_sizes, int n_in,
                              void* d_out, int out_size, void* d_ws, size_t ws_size,
                              hipStream_t stream) {
    const float* x   = (const float*)d_in[0];
    const float* eW1 = (const float*)d_in[1];
    const float* eb1 = (const float*)d_in[2];
    const float* eW2 = (const float*)d_in[3];
    const float* eb2 = (const float*)d_in[4];
    const float* dW1 = (const float*)d_in[5];
    const float* db1 = (const float*)d_in[6];
    const float* dW2 = (const float*)d_in[7];
    const float* db2 = (const float*)d_in[8];
    const float* C0  = (const float*)d_in[9];
    const float* C1  = (const float*)d_in[10];
    const float* C2  = (const float*)d_in[11];

    char* w = (char*)d_ws;
    size_t off = 0;
    auto take = [&](size_t bytes) { void* p = w + off; off = (off + bytes + 255) & ~(size_t)255; return p; };

    bf16_t* H      = (bf16_t*)take((size_t)N_ROWS * D_H * 2);   // H1 then reused as H2
    float*  resid  = (float*) take((size_t)N_ROWS * D_Z * 4);
    bf16_t* rbf    = (bf16_t*)take((size_t)N_ROWS * D_Z * 2);
    float*  outq_f = (float*) take((size_t)N_ROWS * D_Z * 4);
    bf16_t* outq_b = (bf16_t*)take((size_t)N_ROWS * D_Z * 2);
    bf16_t* eW1t   = (bf16_t*)take((size_t)D_IN * D_H * 2);
    bf16_t* eW2t   = (bf16_t*)take((size_t)D_H * D_Z * 2);
    bf16_t* dW1t   = (bf16_t*)take((size_t)D_Z * D_H * 2);
    bf16_t* dW2t   = (bf16_t*)take((size_t)D_H * D_IN * 2);
    bf16_t* C0b    = (bf16_t*)take((size_t)1024 * D_Z * 2);
    bf16_t* C1b    = (bf16_t*)take((size_t)512 * D_Z * 2);
    bf16_t* C2b    = (bf16_t*)take((size_t)256 * D_Z * 2);
    float*  cnorm  = (float*) take(1792 * 4);
    unsigned long long* minbuf = (unsigned long long*)take((size_t)N_ROWS * 8);
    float*  loss   = (float*) take(8 * 4);                      // [c0,c1,c2,recon]

    const int T = 256;
    auto blocks = [](long n, int t) { return (unsigned)((n + t - 1) / t); };

    // weight / codebook conversion (transposed so B fragments load contiguously)
    k_cvt_transpose<<<blocks((long)D_IN * D_H, T), T, 0, stream>>>(eW1, eW1t, D_IN, D_H);
    k_cvt_transpose<<<blocks((long)D_H * D_Z, T), T, 0, stream>>>(eW2, eW2t, D_H, D_Z);
    k_cvt_transpose<<<blocks((long)D_Z * D_H, T), T, 0, stream>>>(dW1, dW1t, D_Z, D_H);
    k_cvt_transpose<<<blocks((long)D_H * D_IN, T), T, 0, stream>>>(dW2, dW2t, D_H, D_IN);
    k_cvt_copy<<<blocks((long)1024 * D_Z, T), T, 0, stream>>>(C0, C0b, (long)1024 * D_Z);
    k_cvt_copy<<<blocks((long)512 * D_Z, T), T, 0, stream>>>(C1, C1b, (long)512 * D_Z);
    k_cvt_copy<<<blocks((long)256 * D_Z, T), T, 0, stream>>>(C2, C2b, (long)256 * D_Z);
    k_cnorm<<<1024, 256, 0, stream>>>(C0, cnorm);
    k_cnorm<<<512, 256, 0, stream>>>(C1, cnorm + 1024);
    k_cnorm<<<256, 256, 0, stream>>>(C2, cnorm + 1536);

    k_fill_f32<<<blocks((long)N_ROWS * D_Z, T), T, 0, stream>>>(outq_f, 0.0f, (size_t)N_ROWS * D_Z);
    k_fill_f32<<<1, 8, 0, stream>>>(loss, 0.0f, 8);

    dim3 blk(256);
    const unsigned GX = N_ROWS / 256;   // 8 waves x 32 rows per block

    // encoder: H1 = GELU(x @ eW1 + eb1)
    k_gemm_f32A_bias_gelu<<<dim3(GX, D_H / 64), blk, 0, stream>>>(
        x, D_IN, eW1t, D_IN, eb1, H, D_H, D_IN);
    // z = H1 @ eW2 + eb2 (f32 residual + bf16 copy)
    k_gemm_bias_dual<<<dim3(GX, D_Z / 64), blk, 0, stream>>>(
        H, D_H, eW2t, D_H, eb2, resid, rbf, D_Z, D_H);

    // RQ stages
    const float* Cs[3] = {C0, C1, C2};
    const bf16_t* Cbs[3] = {C0b, C1b, C2b};
    const int codes[3] = {1024, 512, 256};
    const int cnoff[3] = {0, 1024, 1536};
    for (int s = 0; s < 3; s++) {
        k_fill_u64<<<blocks(N_ROWS, T), T, 0, stream>>>(minbuf, ~0ull, (size_t)N_ROWS);
        k_score_argmin<<<dim3(GX, codes[s] / 64), blk, 0, stream>>>(
            rbf, D_Z, Cbs[s], D_Z, cnorm + cnoff[s], minbuf, D_Z);
        k_quant_update<<<N_ROWS, 256, 0, stream>>>(
            Cs[s], minbuf, resid, rbf, outq_f, outq_b, loss + s);
    }

    // decoder: H2 = GELU(outq @ dW1 + db1)  (reuse H buffer)
    k_gemm_bias_gelu<<<dim3(GX, D_H / 64), blk, 0, stream>>>(
        outq_b, D_Z, dW1t, D_Z, db1, H, D_H, D_Z);
    // out = H2 @ dW2 + db2, fused recon loss vs x
    k_gemm_recon<<<dim3(GX, D_IN / 64), blk, 0, stream>>>(
        H, D_H, dW2t, D_H, db2, x, D_IN, loss + 3, D_H);

    k_finalize<<<1, 1, 0, stream>>>(loss, (float*)d_out);
}
